// CrossAttention_42399917146626
// MI455X (gfx1250) — matrix operand order repaired and checked by reading, not compiled, not run on hardware
//
#include <hip/hip_runtime.h>
#include <hip/hip_bf16.h>

// ---------------------------------------------------------------------------
// CDNA5 (gfx1250) cross-attention: all matmuls through v_wmma_f32_16x16x32_bf16
// Double-buffered LDS, b128 staging everywhere except small weight tiles,
// transposed-V so the big attn*V GEMM streams both operands contiguously.
// ---------------------------------------------------------------------------

typedef __bf16 bf16_t;
typedef __attribute__((ext_vector_type(16))) __bf16 v16bf;
typedef __attribute__((ext_vector_type(8)))  __bf16 v8bf;
typedef __attribute__((ext_vector_type(8)))  float  v8f;
typedef __attribute__((ext_vector_type(4)))  float  f32x4;

__device__ __forceinline__ bf16_t cvt_bf16(float x)  { return (bf16_t)x; }
__device__ __forceinline__ bf16_t cvt_bf16(bf16_t x) { return x; }

__device__ __forceinline__ void store_out(float* p, float v)  { *p = v; }
__device__ __forceinline__ void store_out(bf16_t* p, float v) { *p = (bf16_t)v; }

// 8-element vector load from global (16B/32B aligned by construction).
template<typename T>
__device__ __forceinline__ void ldv8(const T* __restrict__ p, T (&r)[8]) {
  if constexpr (sizeof(T) == 4) {
    f32x4 u0 = *(const f32x4*)p;
    f32x4 u1 = *(const f32x4*)(p + 4);
    #pragma unroll
    for (int i = 0; i < 4; ++i) { r[i] = u0[i]; r[i + 4] = u1[i]; }
  } else {
    v8bf u = *(const v8bf*)p;
    #pragma unroll
    for (int i = 0; i < 8; ++i) r[i] = u[i];
  }
}

// Pack 8 staged values to bf16 (v_cvt_pk_bf16_f32 pairs for f32 sources).
template<typename T>
__device__ __forceinline__ v8bf pk8(const T (&r)[8]) {
  v8bf o;
  #pragma unroll
  for (int i = 0; i < 8; ++i) o[i] = cvt_bf16(r[i]);
  return o;
}

// A-fragment (16x32 bf16, ISA 7.12.2): lanes 0-15 hold K={0..7,16..23},
// lanes 16-31 hold K={8..15,24..31}.
__device__ __forceinline__ v16bf load_fragA_lds(const bf16_t* p0) {
  v8bf lo = *(const v8bf*)(p0);        // K = base .. base+7
  v8bf hi = *(const v8bf*)(p0 + 16);   // K = base+16 .. base+23
  return __builtin_shufflevector(lo, hi, 0,1,2,3,4,5,6,7,8,9,10,11,12,13,14,15);
}

// B-fragment (32x16 bf16): lane holds column n=lane%16; lanes 0-15 carry
// K={0..7,16..23}, lanes 16-31 carry K={8..15,24..31} (matches A-fragment).
__device__ __forceinline__ v16bf load_fragB_lds(const bf16_t* p0) {
  v8bf lo = *(const v8bf*)(p0);
  v8bf hi = *(const v8bf*)(p0 + 16);
  return __builtin_shufflevector(lo, hi, 0,1,2,3,4,5,6,7,8,9,10,11,12,13,14,15);
}

// Generic batched GEMM:
//   C[z] = scale * A[z] (MxK row-major, ld=K) x B[z] (+ bias[n])
//   B operand: KxN row-major (ld=ldB) or BTRANS: NxK row-major (ld=ldB)
//   TRANSC: store C transposed, C[n*ldC + m] (vectorized b128 bf16 stores)
// batch offset of X = (z / zdiv) * sX_hi + (z % zdiv) * sX_lo   (elements)
// Requires: M % 64 == 0, N % 128 == 0, K % 32 == 0.
template<typename TA, typename TB, typename TO, bool BTRANS, bool TRANSC, bool HASBIAS>
__global__ __launch_bounds__(256)
void gemm_wmma_bf16(const TA* __restrict__ A, const TB* __restrict__ Bm,
                    TO* __restrict__ C, const float* __restrict__ bias,
                    int M, int N, int K, int ldB, int ldC, int zdiv,
                    long long sA_lo, long long sA_hi,
                    long long sB_lo, long long sB_hi,
                    long long sC_lo, long long sC_hi,
                    float scale)
{
  __shared__ __align__(16) bf16_t At[2][64][40];    // 2 x 64 x 32 (+pad)
  __shared__ __align__(16) bf16_t Bt[2][128][40];   // 2 x Bt[n][k] 128 x 32 (+pad)

  const int tid  = threadIdx.x;
  const int lane = tid & 31;          // wave32
  const int wave = tid >> 5;          // 8 waves
  const int wy   = wave >> 1;         // M sub-tile (0..3)
  const int wx   = wave & 1;          // N half (0..1)

  const int z  = blockIdx.z;
  const int zh = z / zdiv;
  const int zl = z - zh * zdiv;
  A  += zh * sA_hi + zl * sA_lo;
  Bm += zh * sB_hi + zl * sB_lo;
  C  += zh * sC_hi + zl * sC_lo;

  const int bm  = blockIdx.y * 64;
  const int bn  = blockIdx.x * 128;

  // staging coordinates
  const int ar  = tid >> 2;           // 0..63
  const int ac0 = (tid & 3) * 8;      // 0,8,16,24
  const int bnr = tid >> 1;           // 0..127  (BTRANS: rows of NxK source)
  const int bc0 = (tid & 1) * 16;     // 0,16
  const int bk  = tid >> 3;           // 0..31   (!BTRANS: rows of KxN source)
  const int bn0 = (tid & 7) * 16;     // 0..112

  TA aR[8];
  TB bR0[8], bR1[8];

  auto fetch = [&](int k0) {
    ldv8(A + (long long)(bm + ar) * K + (k0 + ac0), aR);
    if constexpr (BTRANS) {
      const TB* bp = Bm + (long long)(bn + bnr) * ldB + (k0 + bc0);
      ldv8(bp,     bR0);
      ldv8(bp + 8, bR1);
    } else {
      const TB* bp = Bm + (long long)(k0 + bk) * ldB + (bn + bn0);
      ldv8(bp,     bR0);
      ldv8(bp + 8, bR1);
    }
  };

  auto stage = [&](int buf) {
    *(v8bf*)&At[buf][ar][ac0] = pk8(aR);
    if constexpr (BTRANS) {
      *(v8bf*)&Bt[buf][bnr][bc0]     = pk8(bR0);
      *(v8bf*)&Bt[buf][bnr][bc0 + 8] = pk8(bR1);
    } else {
      #pragma unroll
      for (int i = 0; i < 8; ++i) Bt[buf][bn0 + i][bk]     = cvt_bf16(bR0[i]);
      #pragma unroll
      for (int i = 0; i < 8; ++i) Bt[buf][bn0 + 8 + i][bk] = cvt_bf16(bR1[i]);
    }
  };

  v8f acc[4] = {};

  fetch(0);
  stage(0);
  __syncthreads();

  int buf = 0;
  for (int k0 = 0; k0 < K; k0 += 32) {
    const bool more = (k0 + 32) < K;
    // Prefetch next-next A tile into L2 (global_prefetch_b8).
    if (k0 + 64 < K)
      __builtin_prefetch(A + (long long)(bm + ar) * K + (k0 + 64 + ac0), 0, 1);
    // Issue next tile's global loads so they overlap the WMMAs below.
    if (more) fetch(k0 + 32);

    // Hoist all fragments, then issue the 4 WMMAs back-to-back.
    const int abase = (lane >= 16) ? 8 : 0;
    const int bbase = (lane >= 16) ? 8 : 0;
    const int rsel  = lane & 15;
    v16bf af  = load_fragA_lds(&At[buf][wy * 16 + rsel][abase]);
    v16bf bf0 = load_fragB_lds(&Bt[buf][wx * 64 +  0 + rsel][bbase]);
    v16bf bf1 = load_fragB_lds(&Bt[buf][wx * 64 + 16 + rsel][bbase]);
    v16bf bf2 = load_fragB_lds(&Bt[buf][wx * 64 + 32 + rsel][bbase]);
    v16bf bf3 = load_fragB_lds(&Bt[buf][wx * 64 + 48 + rsel][bbase]);
    acc[0] = __builtin_amdgcn_wmma_f32_16x16x32_bf16(false, af, false, bf0, (short)0, acc[0], false, false);
    acc[1] = __builtin_amdgcn_wmma_f32_16x16x32_bf16(false, af, false, bf1, (short)0, acc[1], false, false);
    acc[2] = __builtin_amdgcn_wmma_f32_16x16x32_bf16(false, af, false, bf2, (short)0, acc[2], false, false);
    acc[3] = __builtin_amdgcn_wmma_f32_16x16x32_bf16(false, af, false, bf3, (short)0, acc[3], false, false);

    if (more) {
      __syncthreads();        // all reads of LDS[buf^1] from prev iter done
      stage(buf ^ 1);
      __syncthreads();        // stores visible before next iter's reads
      buf ^= 1;
    }
  }

  // Epilogue: C/D layout (ISA 7.12.2): elem i -> row m0+i, col = lane%16.
  const int m0    = bm + wy * 16 + ((lane >= 16) ? 8 : 0);
  const int nlane = lane & 15;
  #pragma unroll
  for (int j = 0; j < 4; ++j) {
    const int n  = bn + wx * 64 + j * 16 + nlane;
    const float bv = HASBIAS ? bias[n] : 0.0f;
    if constexpr (TRANSC && sizeof(TO) == 2) {
      // Lane holds 8 consecutive rows of one column -> one b128 store of C^T.
      v8bf o;
      #pragma unroll
      for (int i = 0; i < 8; ++i) o[i] = (bf16_t)(acc[j][i] * scale + bv);
      *(v8bf*)&C[(long long)n * ldC + m0] = o;
    } else if constexpr (TRANSC) {
      f32x4 o0, o1;
      #pragma unroll
      for (int i = 0; i < 4; ++i) { o0[i] = acc[j][i] * scale + bv;
                                    o1[i] = acc[j][i + 4] * scale + bv; }
      *(f32x4*)&C[(long long)n * ldC + m0]     = o0;
      *(f32x4*)&C[(long long)n * ldC + m0 + 4] = o1;
    } else {
      #pragma unroll
      for (int i = 0; i < 8; ++i) {
        float v = acc[j][i] * scale + bv;
        store_out(&C[(long long)(m0 + i) * ldC + n], v);
      }
    }
  }
}

// Row softmax in place for rows of exactly 1024 floats: one f32x4 per thread.
__global__ __launch_bounds__(256)
void softmax_rows_1024(float* __restrict__ attn)
{
  const int tid = threadIdx.x;
  f32x4* p = (f32x4*)(attn + (long long)blockIdx.x * 1024);
  f32x4 v = p[tid];

  __shared__ float sred[8];
  __shared__ float sbc;

  // 1) row max
  float m = fmaxf(fmaxf(v[0], v[1]), fmaxf(v[2], v[3]));
  #pragma unroll
  for (int o = 16; o > 0; o >>= 1) m = fmaxf(m, __shfl_xor(m, o, 32));
  if ((tid & 31) == 0) sred[tid >> 5] = m;
  __syncthreads();
  if (tid == 0) {
    float t = sred[0];
    #pragma unroll
    for (int i = 1; i < 8; ++i) t = fmaxf(t, sred[i]);
    sbc = t;
  }
  __syncthreads();
  m = sbc;

  // 2) exp + sum
  f32x4 e;
  #pragma unroll
  for (int i = 0; i < 4; ++i) e[i] = __expf(v[i] - m);
  float s = (e[0] + e[1]) + (e[2] + e[3]);
  #pragma unroll
  for (int o = 16; o > 0; o >>= 1) s += __shfl_xor(s, o, 32);
  if ((tid & 31) == 0) sred[tid >> 5] = s;
  __syncthreads();
  if (tid == 0) {
    float t = 0.0f;
    #pragma unroll
    for (int i = 0; i < 8; ++i) t += sred[i];
    sbc = t;
  }
  __syncthreads();

  // 3) normalize + store
  const float inv = 1.0f / sbc;
  #pragma unroll
  for (int i = 0; i < 4; ++i) e[i] *= inv;
  p[tid] = e;
}

// ---------------------------------------------------------------------------
// Host launcher
// ---------------------------------------------------------------------------
extern "C" void kernel_launch(void* const* d_in, const int* in_sizes, int n_in,
                              void* d_out, int out_size, void* d_ws, size_t ws_size,
                              hipStream_t stream) {
  (void)in_sizes; (void)n_in; (void)out_size; (void)ws_size;

  constexpr int Bn = 8, LQ = 1024, LKV = 1024, EQ = 1024, EKV = 768, H = 8, HD = 128;
  constexpr long long MQ = (long long)Bn * LQ;          // 8192 rows of q/k/v matrices

  const float* query  = (const float*)d_in[0];
  const float* key    = (const float*)d_in[1];
  const float* value  = (const float*)d_in[2];
  const float* w_q    = (const float*)d_in[3];
  const float* w_k    = (const float*)d_in[4];
  const float* w_v    = (const float*)d_in[5];
  const float* proj_w = (const float*)d_in[6];
  const float* proj_b = (const float*)d_in[7];

  float* out  = (float*)d_out;                          // [B, LQ, EQ]
  float* attn = out + (long long)Bn * LQ * EQ;          // [H*B, LQ, LKV]

  // Workspace: 3 x (H*B*L*HD) bf16 buffers = 48 MB.
  // q_ws is reused for the head-concat output (q is dead after the score GEMM).
  bf16_t* q_ws   = (bf16_t*)d_ws;                       // [H][B*LQ][HD]
  bf16_t* k_ws   = q_ws + (long long)H * MQ * HD;       // [H][B*LKV][HD]
  bf16_t* vT_ws  = k_ws + (long long)H * MQ * HD;       // [H][HD][B*LKV]  (transposed)
  bf16_t* concat = q_ws;                                // [B*LQ][H*HD]

  const dim3 blk(256);
  const float inv_sqrt_hd = 0.08838834764831845f;       // 1/sqrt(128)

  // 1) q[h] = query (8192x1024) x w_q[h] (1024x128)            -> bf16
  gemm_wmma_bf16<float, float, bf16_t, false, false, false>
      <<<dim3(1, 128, H), blk, 0, stream>>>(
      query, w_q, q_ws, nullptr,
      (int)MQ, HD, EQ, HD, HD, 1,
      0, 0,
      0, (long long)EQ * HD,
      0, MQ * HD,
      1.0f);

  // 2) k[h] = key (8192x768) x w_k[h] (768x128)                -> bf16
  gemm_wmma_bf16<float, float, bf16_t, false, false, false>
      <<<dim3(1, 128, H), blk, 0, stream>>>(
      key, w_k, k_ws, nullptr,
      (int)MQ, HD, EKV, HD, HD, 1,
      0, 0,
      0, (long long)EKV * HD,
      0, MQ * HD,
      1.0f);

  // 3) vT[h] = (value x w_v[h])^T : TRANSC epilogue            -> bf16 [HD][8192]
  gemm_wmma_bf16<float, float, bf16_t, false, true, false>
      <<<dim3(1, 128, H), blk, 0, stream>>>(
      value, w_v, vT_ws, nullptr,
      (int)MQ, HD, EKV, HD, (int)MQ, 1,
      0, 0,
      0, (long long)EKV * HD,
      0, (long long)HD * MQ,
      1.0f);

  // 4) scores[z=h*B+b] = q_hb (1024x128) x k_hb^T * 1/sqrt(HD) -> f32 attn buf
  gemm_wmma_bf16<bf16_t, bf16_t, float, true, false, false>
      <<<dim3(8, 16, H * Bn), blk, 0, stream>>>(
      q_ws, k_ws, attn, nullptr,
      LQ, LKV, HD, HD, LKV, 1,
      0, (long long)LQ * HD,
      0, (long long)LKV * HD,
      0, (long long)LQ * LKV,
      inv_sqrt_hd);

  // 5) softmax over each of the H*B*LQ rows (in place, f32)
  softmax_rows_1024<<<dim3(H * Bn * LQ), blk, 0, stream>>>(attn);

  // 6) head_out[z=h*B+b] = attn_hb (1024x1024) x v_hb (1024x128) -> bf16 concat
  //    B operand is vT (NxK = HD x LKV slices, ld = 8192) -> pure b128 staging.
  gemm_wmma_bf16<float, bf16_t, bf16_t, true, false, false>
      <<<dim3(1, 16, H * Bn), blk, 0, stream>>>(
      attn, vT_ws, concat, nullptr,
      LQ, HD, LKV, (int)MQ, H * HD, Bn,
      (long long)LQ * LKV, (long long)Bn * LQ * LKV,   // attn: lo per-b, hi per-h
      (long long)LQ,       (long long)HD * MQ,         // vT:   lo per-b, hi per-h
      (long long)LQ * H * HD, (long long)HD,           // C:    lo per-b, hi per-h
      1.0f);

  // 7) out = concat (8192x1024) x proj_w^T (+ proj_b)          -> f32
  gemm_wmma_bf16<bf16_t, float, float, true, false, true>
      <<<dim3(8, 128, 1), blk, 0, stream>>>(
      concat, proj_w, out, proj_b,
      (int)MQ, EQ, H * HD, H * HD, EQ, 1,
      0, 0,
      0, 0,
      0, 0,
      1.0f);
}
